// GEDNet_61735859913576
// MI455X (gfx1250) — matrix operand order repaired and checked by reading, not compiled, not run on hardware
//
#include <hip/hip_runtime.h>
#include <hip/hip_bf16.h>
#include <math.h>

// ---------------------------------------------------------------------------
// Types for CDNA5 WMMA (wave32): 16x16x32 bf16 -> f32 accumulate
// ---------------------------------------------------------------------------
typedef __bf16 bf16_t;
typedef __attribute__((ext_vector_type(16))) __bf16 v16bf;
typedef __attribute__((ext_vector_type(8)))  float  v8f;
typedef int v4i_vec __attribute__((vector_size(16)));   // matches builtin param

#define BM 128
#define BN 64
#define BK 32
#define LDS_K (BK + 8)   // 40 bf16 = 80B row stride (16B aligned, bank-skewed)

// --- gfx1250 async global->LDS staging (probe-confirmed: 4 args, v4i ptrs) --
#if defined(__AMDGCN__) && __has_builtin(__builtin_amdgcn_global_load_async_to_lds_b128)
#define HAVE_ASYNC_LDS 1
#define ASYNC_B128(g, l)                                                       \
    __builtin_amdgcn_global_load_async_to_lds_b128(                            \
        (__attribute__((address_space(1))) v4i_vec*)(uintptr_t)(g),            \
        (__attribute__((address_space(3))) v4i_vec*)(unsigned)(uintptr_t)(l),  \
        0, 0)
#else
#define HAVE_ASYNC_LDS 0
#endif

__device__ __forceinline__ void wait_async_zero()
{
#if defined(__AMDGCN__) && __has_builtin(__builtin_amdgcn_s_wait_asynccnt)
    __builtin_amdgcn_s_wait_asynccnt(0);
#elif defined(__AMDGCN__)
    asm volatile("s_wait_asynccnt 0" ::: "memory");
#endif
}

__device__ __forceinline__ int imin(int a, int b) { return a < b ? a : b; }

// Fragment builders per CDNA5 ISA 7.12.2 (wave32, 16-bit):
//  A 16x32: lane = m + 16*((k&8)!=0); vgpr = (k>>4)*4 + ((k&7)>>1)
//  B 32x16: lane = n + 16*((k&8)!=0); vgpr = (k>>4)*4 + ((k&7)>>1)
__device__ __forceinline__ v16bf frag_a(const bf16_t* __restrict__ row, int lhi)
{
    v16bf f;
    #pragma unroll
    for (int v = 0; v < 8; ++v) {
        const int ka = (v >> 2) * 16 + lhi * 8 + (v & 3) * 2;
        f[2 * v]     = row[ka];
        f[2 * v + 1] = row[ka + 1];
    }
    return f;
}
__device__ __forceinline__ v16bf frag_b(const bf16_t* __restrict__ row, int lhi)
{
    v16bf f;
    #pragma unroll
    for (int v = 0; v < 8; ++v) {
        const int kb = (v >> 2) * 16 + lhi * 8 + (v & 3) * 2;
        f[2 * v]     = row[kb];
        f[2 * v + 1] = row[kb + 1];
    }
    return f;
}

// ---------------------------------------------------------------------------
// GEMM: C[M,N] = act( A[M,K] * B[N,K]^T + bias )    (NT, bf16 in / f32 acc)
// REQUIREMENT: K % 32 == 0 and A/B rows zero-padded to K (all producers below
// guarantee this), so staging has NO per-element guards: rows are clamped.
// 256 threads = 8 waves (4x2); each wave computes a 32x32 C tile = 4 v_wmma
// per K-step. LDS is double-buffered; tiles staged with async b128 loads.
// ---------------------------------------------------------------------------
__global__ __launch_bounds__(256)
void k_gemm_bf16_nt_wmma(const bf16_t* __restrict__ A, int lda,
                         const bf16_t* __restrict__ B, int ldb,
                         float* __restrict__ C, bf16_t* __restrict__ Cb, int ldc,
                         const float* __restrict__ bias,
                         int M, int N, int K, int act)
{
    __shared__ bf16_t sA[2][BM * LDS_K];
    __shared__ bf16_t sB[2][BN * LDS_K];

    const int tid  = threadIdx.x;
    const int lane = tid & 31;
    const int wave = tid >> 5;
    const int m0   = blockIdx.y * BM;
    const int n0   = blockIdx.x * BN;
    const int wm   = (wave >> 1) * 32;   // 0,32,64,96
    const int wn   = (wave & 1) * 32;    // 0,32
    const int lm   = lane & 15;
    const int lhi  = lane >> 4;

    // staging assignment (no guards: clamped rows, K padded)
    const int arow = tid >> 1;                    // 0..127
    const int acol = (tid & 1) * 16;              // 0,16  (bf16 units)
    const int brow = tid >> 2;                    // 0..63
    const int bcol = (tid & 3) * 8;               // 0,8,16,24
    const size_t am = (size_t)imin(m0 + arow, M - 1);
    const size_t bn = (size_t)imin(n0 + brow, N - 1);

    auto stage = [&](int kk, int sbuf) {
        const bf16_t* ap = A + am * lda + kk + acol;
        const bf16_t* bp = B + bn * ldb + kk + bcol;
        bf16_t* la = &sA[sbuf][arow * LDS_K + acol];
        bf16_t* lb = &sB[sbuf][brow * LDS_K + bcol];
#if HAVE_ASYNC_LDS
        ASYNC_B128(ap,     la);
        ASYNC_B128(ap + 8, la + 8);
        ASYNC_B128(bp,     lb);
#else
        *(uint4*)la       = *(const uint4*)ap;
        *(uint4*)(la + 8) = *(const uint4*)(ap + 8);
        *(uint4*)lb       = *(const uint4*)bp;
#endif
    };

    v8f acc[2][2] = {};

    stage(0, 0);
    wait_async_zero();
    __syncthreads();

    int buf = 0;
    for (int k0 = 0; k0 < K; k0 += BK) {
        const bool more = (k0 + BK) < K;
        if (more) stage(k0 + BK, buf ^ 1);        // prefetch next tile (async)

        const v16bf a0 = frag_a(&sA[buf][(wm + lm) * LDS_K], lhi);
        const v16bf a1 = frag_a(&sA[buf][(wm + 16 + lm) * LDS_K], lhi);
        const v16bf b0 = frag_b(&sB[buf][(wn + lm) * LDS_K], lhi);
        const v16bf b1 = frag_b(&sB[buf][(wn + 16 + lm) * LDS_K], lhi);

        acc[0][0] = __builtin_amdgcn_wmma_f32_16x16x32_bf16(
                        false, a0, false, b0, (short)0, acc[0][0], false, false);
        acc[0][1] = __builtin_amdgcn_wmma_f32_16x16x32_bf16(
                        false, a0, false, b1, (short)0, acc[0][1], false, false);
        acc[1][0] = __builtin_amdgcn_wmma_f32_16x16x32_bf16(
                        false, a1, false, b0, (short)0, acc[1][0], false, false);
        acc[1][1] = __builtin_amdgcn_wmma_f32_16x16x32_bf16(
                        false, a1, false, b1, (short)0, acc[1][1], false, false);

        if (more) wait_async_zero();
        __syncthreads();
        buf ^= 1;
    }

    // epilogue: bias + activation + f32 / padded-bf16 stores
    #pragma unroll
    for (int i = 0; i < 2; ++i) {
        #pragma unroll
        for (int j = 0; j < 2; ++j) {
            #pragma unroll
            for (int v = 0; v < 8; ++v) {
                const int m = m0 + wm + i * 16 + v + 8 * lhi;
                const int n = n0 + wn + j * 16 + lm;
                if (m < M && n < N) {
                    float r = acc[i][j][v];
                    if (bias) r += bias[n];
                    if (act == 1)      r = fmaxf(r, 0.0f);
                    else if (act == 2) r = 0.5f * r * (1.0f + erff(r * 0.70710678f));
                    if (C)  C [(size_t)m * ldc + n] = r;
                    if (Cb) Cb[(size_t)m * ldc + n] = (bf16_t)r;
                }
            }
        }
    }
}

// ---------------------------------------------------------------------------
// Producers — every bf16 GEMM operand is written with rows padded to Cp
// (Cp % 32 == 0, padding zero-filled) so the GEMM needs no K guards.
// ---------------------------------------------------------------------------
__global__ void k_f32_to_bf16_rows(const float* __restrict__ x, bf16_t* __restrict__ y,
                                   size_t R, int C, int Cp)
{
    const size_t n = R * (size_t)Cp;
    for (size_t i = blockIdx.x * (size_t)blockDim.x + threadIdx.x; i < n;
         i += (size_t)gridDim.x * blockDim.x) {
        const int c = i % Cp;
        const size_t r = i / Cp;
        y[i] = (c < C) ? (bf16_t)x[r * C + c] : (bf16_t)0.0f;
    }
}

__global__ void k_nchw_to_nhwc_bf16(const float* __restrict__ x, bf16_t* __restrict__ y,
                                    int B, int C, int H, int W)
{
    const size_t n = (size_t)B * C * H * W;
    for (size_t i = blockIdx.x * (size_t)blockDim.x + threadIdx.x; i < n;
         i += (size_t)gridDim.x * blockDim.x) {
        const int c = i % C;
        size_t p = i / C;
        const int wx = p % W; p /= W;
        const int hy = p % H; const int b = p / H;
        y[i] = (bf16_t)x[(((size_t)b * C + c) * H + hy) * W + wx];
    }
}

__global__ void k_nhwc_to_nchw_f32(const float* __restrict__ x, float* __restrict__ y,
                                   int B, int C, int H, int W)
{
    const size_t n = (size_t)B * C * H * W;
    for (size_t i = blockIdx.x * (size_t)blockDim.x + threadIdx.x; i < n;
         i += (size_t)gridDim.x * blockDim.x) {
        const int wx = i % W;
        size_t p = i / W;
        const int hy = p % H; p /= H;
        const int c = p % C; const int b = p / C;
        y[i] = x[(((size_t)b * H + hy) * W + wx) * C + c];
    }
}

// im2col 3x3, NHWC in; col[p, c*9+ky*3+kx] with row stride Kp (zero padded)
__global__ void k_im2col3x3_bf16(const bf16_t* __restrict__ x, bf16_t* __restrict__ col,
                                 int B, int H, int W, int C, int stride, int pad,
                                 int OH, int OW, int Kp)
{
    const size_t total = (size_t)B * OH * OW * Kp;
    for (size_t i = blockIdx.x * (size_t)blockDim.x + threadIdx.x; i < total;
         i += (size_t)gridDim.x * blockDim.x) {
        const int kk = i % Kp;
        const size_t p = i / Kp;
        bf16_t v = (bf16_t)0.0f;
        if (kk < 9 * C) {
            const int c = kk / 9, t = kk % 9;
            const int kx = t % 3, ky = t / 3;
            const int ox = p % OW;
            size_t q = p / OW;
            const int oy = q % OH;
            const int b  = q / OH;
            const int ix = ox * stride - pad + kx;
            const int iy = oy * stride - pad + ky;
            if (ix >= 0 && ix < W && iy >= 0 && iy < H)
                v = x[(((size_t)b * H + iy) * W + ix) * C + c];
        }
        col[i] = v;
    }
}

// ---------------------------------------------------------------------------
// BatchNorm (batch statistics) — NHWC: x is [P, C]
// ---------------------------------------------------------------------------
__global__ __launch_bounds__(256)
void k_channel_stats(const float* __restrict__ x, int P, int C,
                     float* __restrict__ mean, float* __restrict__ var)
{
    __shared__ float s1[256], s2[256];
    const int c = blockIdx.x;
    float a = 0.f, b = 0.f;
    for (int p = threadIdx.x; p < P; p += 256) {
        const float v = x[(size_t)p * C + c];
        a += v; b += v * v;
    }
    s1[threadIdx.x] = a; s2[threadIdx.x] = b;
    __syncthreads();
    for (int s = 128; s > 0; s >>= 1) {
        if ((int)threadIdx.x < s) {
            s1[threadIdx.x] += s1[threadIdx.x + s];
            s2[threadIdx.x] += s2[threadIdx.x + s];
        }
        __syncthreads();
    }
    if (threadIdx.x == 0) {
        const float m = s1[0] / (float)P;
        mean[c] = m;
        var[c]  = fmaxf(s2[0] / (float)P - m * m, 0.0f);
    }
}

// y_bf16 written with padded row stride Cp (zero pad); y_f32 unpadded
__global__ void k_bn_act(const float* __restrict__ x, int P, int C, int Cp,
                         const float* __restrict__ mean, const float* __restrict__ var,
                         const float* __restrict__ g, const float* __restrict__ bta,
                         int act, float* __restrict__ yf, bf16_t* __restrict__ yb)
{
    const size_t n = (size_t)P * Cp;
    for (size_t i = blockIdx.x * (size_t)blockDim.x + threadIdx.x; i < n;
         i += (size_t)gridDim.x * blockDim.x) {
        const int c = i % Cp;
        const size_t p = i / Cp;
        if (c < C) {
            float r = (x[p * C + c] - mean[c]) * rsqrtf(var[c] + 1e-5f) * g[c] + bta[c];
            if (act == 1)      r = fmaxf(r, 0.0f);
            else if (act == 2) r = 0.5f * r * (1.0f + erff(r * 0.70710678f));
            if (yf) yf[p * C + c] = r;
            if (yb) yb[i] = (bf16_t)r;
        } else if (yb) {
            yb[i] = (bf16_t)0.0f;
        }
    }
}

// ---------------------------------------------------------------------------
// Pooling / resize / misc elementwise (NHWC)
// ---------------------------------------------------------------------------
__global__ void k_maxpool2_nhwc(const float* __restrict__ x, float* __restrict__ y,
                                int B, int H, int W, int C)
{
    const int OH = H / 2, OW = W / 2;
    const size_t n = (size_t)B * OH * OW * C;
    for (size_t i = blockIdx.x * (size_t)blockDim.x + threadIdx.x; i < n;
         i += (size_t)gridDim.x * blockDim.x) {
        const int c = i % C;
        size_t p = i / C;
        const int ox = p % OW; p /= OW;
        const int oy = p % OH; const int b = p / OH;
        const size_t base = (((size_t)b * H + oy * 2) * W + ox * 2) * C + c;
        float m = x[base];
        m = fmaxf(m, x[base + C]);
        m = fmaxf(m, x[base + (size_t)W * C]);
        m = fmaxf(m, x[base + (size_t)W * C + C]);
        y[i] = m;
    }
}

__global__ void k_avgpool_nhwc(const float* __restrict__ x, float* __restrict__ y,
                               int B, int H, int W, int C, int r)
{
    const int OH = H / r, OW = W / r;
    const size_t n = (size_t)B * OH * OW * C;
    for (size_t i = blockIdx.x * (size_t)blockDim.x + threadIdx.x; i < n;
         i += (size_t)gridDim.x * blockDim.x) {
        const int c = i % C;
        size_t p = i / C;
        const int ox = p % OW; p /= OW;
        const int oy = p % OH; const int b = p / OH;
        float s = 0.f;
        for (int dy = 0; dy < r; ++dy)
            for (int dx = 0; dx < r; ++dx)
                s += x[(((size_t)b * H + oy * r + dy) * W + ox * r + dx) * C + c];
        y[i] = s / (float)(r * r);
    }
}

__global__ void k_up2_bilinear_nhwc(const float* __restrict__ x, float* __restrict__ y,
                                    int B, int H, int W, int C)
{
    const int OH = 2 * H, OW = 2 * W;
    const size_t n = (size_t)B * OH * OW * C;
    for (size_t i = blockIdx.x * (size_t)blockDim.x + threadIdx.x; i < n;
         i += (size_t)gridDim.x * blockDim.x) {
        const int c = i % C;
        size_t p = i / C;
        const int ox = p % OW; p /= OW;
        const int oy = p % OH; const int b = p / OH;
        const float sx = 0.5f * (ox + 0.5f) - 0.5f;
        const float sy = 0.5f * (oy + 0.5f) - 0.5f;
        const int x0i = (int)floorf(sx), y0i = (int)floorf(sy);
        const float fx = sx - x0i, fy = sy - y0i;
        const int x0c = min(max(x0i, 0), W - 1), x1c = min(max(x0i + 1, 0), W - 1);
        const int y0c = min(max(y0i, 0), H - 1), y1c = min(max(y0i + 1, 0), H - 1);
        const size_t bb = (size_t)b * H;
        const float v00 = x[((bb + y0c) * W + x0c) * C + c];
        const float v01 = x[((bb + y0c) * W + x1c) * C + c];
        const float v10 = x[((bb + y1c) * W + x0c) * C + c];
        const float v11 = x[((bb + y1c) * W + x1c) * C + c];
        y[i] = (1.f - fy) * ((1.f - fx) * v00 + fx * v01) +
               fy * ((1.f - fx) * v10 + fx * v11);
    }
}

__global__ void k_add_f32(const float* __restrict__ a, const float* __restrict__ b,
                          float* __restrict__ o, size_t n)
{
    for (size_t i = blockIdx.x * (size_t)blockDim.x + threadIdx.x; i < n;
         i += (size_t)gridDim.x * blockDim.x)
        o[i] = a[i] + b[i];
}

__global__ void k_add_pos_nhwc(float* __restrict__ x, const float* __restrict__ pos,
                               int B, int H, int W, int C)
{
    const size_t n = (size_t)B * H * W * C;
    for (size_t i = blockIdx.x * (size_t)blockDim.x + threadIdx.x; i < n;
         i += (size_t)gridDim.x * blockDim.x) {
        const int c  = i % C;
        const size_t hw = (i / C) % ((size_t)H * W);
        x[i] += pos[(size_t)c * H * W + hw];
    }
}

__global__ void k_copy_ch(const float* __restrict__ src, int P, int Cs,
                          float* __restrict__ dst, int Cd, int coff)
{
    const size_t n = (size_t)P * Cs;
    for (size_t i = blockIdx.x * (size_t)blockDim.x + threadIdx.x; i < n;
         i += (size_t)gridDim.x * blockDim.x) {
        const int c = i % Cs;
        const size_t p = i / Cs;
        dst[p * Cd + coff + c] = src[i];
    }
}

// ---------------------------------------------------------------------------
// Grapher pieces
// ---------------------------------------------------------------------------
__global__ void k_rownorm_bf16(const float* __restrict__ x, int P, int C, int Cp,
                               bf16_t* __restrict__ y)
{
    for (int p = blockIdx.x * blockDim.x + threadIdx.x; p < P;
         p += gridDim.x * blockDim.x) {
        float s = 0.f;
        for (int c = 0; c < C; ++c) { const float v = x[(size_t)p * C + c]; s += v * v; }
        const float inv = 1.0f / fmaxf(sqrtf(s), 1e-12f);
        for (int c = 0; c < Cp; ++c)
            y[(size_t)p * Cp + c] =
                (c < C) ? (bf16_t)(x[(size_t)p * C + c] * inv) : (bf16_t)0.0f;
    }
}

// dist = 2 - 2*ip + rel ; per-row top-(k*dil) smallest, emit every dil-th index
__global__ void k_topk_rows(const float* __restrict__ ip, const float* __restrict__ rel,
                            int N, int Mcol, int kd, int dil, int kout,
                            int* __restrict__ idx)
{
    for (int r = blockIdx.x * blockDim.x + threadIdx.x; r < N;
         r += gridDim.x * blockDim.x) {
        float bv[32]; int bi[32];
        for (int j = 0; j < kd; ++j) { bv[j] = 3.4e38f; bi[j] = 0; }
        for (int m = 0; m < Mcol; ++m) {
            const float d = 2.f - 2.f * ip[(size_t)r * Mcol + m]
                          + rel[(size_t)r * Mcol + m];
            if (d < bv[kd - 1]) {
                int j = kd - 1;
                while (j > 0 && bv[j - 1] > d) {
                    bv[j] = bv[j - 1]; bi[j] = bi[j - 1]; --j;
                }
                bv[j] = d; bi[j] = m;
            }
        }
        for (int j = 0; j < kout; ++j)
            idx[(size_t)r * kout + j] = bi[j * dil];
    }
}

// z for grouped conv, channels interleaved (x,mr) and padded per group:
// logical channel zc = g*KG + j (j < KG) maps to z[p, g*GP + j]; pad -> 0.
// zc even -> x[p, zc/2]; zc odd -> max_j' (y[idx[p,j'], zc/2] - x[p, zc/2])
__global__ void k_build_z(const float* __restrict__ x, const float* __restrict__ y,
                          const int* __restrict__ idx, int P, int C, int kout,
                          int KG, int GP, int NG, bf16_t* __restrict__ z)
{
    const int SP = NG * GP;
    const size_t n = (size_t)P * SP;
    for (size_t i = blockIdx.x * (size_t)blockDim.x + threadIdx.x; i < n;
         i += (size_t)gridDim.x * blockDim.x) {
        const int rem = i % SP;
        const size_t p = i / SP;
        const int g = rem / GP, j = rem % GP;
        bf16_t out = (bf16_t)0.0f;
        if (j < KG) {
            const int zc = g * KG + j;
            const int c = zc >> 1, s = zc & 1;
            const float xv = x[p * C + c];
            if (s == 0) out = (bf16_t)xv;
            else {
                float mx = -3.4e38f;
                for (int jj = 0; jj < kout; ++jj)
                    mx = fmaxf(mx, y[(size_t)idx[p * kout + jj] * C + c] - xv);
                out = (bf16_t)mx;
            }
        }
        z[i] = out;
    }
}

// ---------------------------------------------------------------------------
// Host-side helpers (all launches on `stream`; graph-capture safe)
// ---------------------------------------------------------------------------
static inline void* wsget(void* base, size_t& off, size_t bytes)
{
    void* p = (char*)base + off;
    off = (off + bytes + 255) & ~(size_t)255;
    return p;
}

static inline int padk(int k) { return (k + 31) & ~31; }

static inline void gemm(hipStream_t s,
                        const bf16_t* A, int lda, const bf16_t* B, int ldb,
                        float* C, bf16_t* Cb, int ldc, const float* bias,
                        int M, int N, int K, int act)
{
    dim3 g((N + BN - 1) / BN, (M + BM - 1) / BM);
    k_gemm_bf16_nt_wmma<<<g, 256, 0, s>>>(A, lda, B, ldb, C, Cb, ldc, bias, M, N, K, act);
}

static inline int gs(size_t n)
{
    size_t b = (n + 255) / 256;
    if (b > 16384) b = 16384;
    if (b < 1) b = 1;
    return (int)b;
}

extern "C" void kernel_launch(void* const* d_in, const int* in_sizes, int n_in,
                              void* d_out, int out_size, void* d_ws, size_t ws_size,
                              hipStream_t stream)
{
    // Flattened setup_inputs order: 0 inputs | 1..8 inc | 9..48 down | 49..60 stem
    // 61 pos_embed | 62.. bb[0][0] {gr fc1 w/b, fc1n g/b, g w/b, gn g/b,
    // fc2 w/b, fc2n g/b, rel}, ffn {fc1 w/b, n1 g/b, fc2 w/b, n2 g/b}
    auto PIN = [&](int i) -> const float* {
        return (i < n_in) ? (const float*)d_in[i] : (const float*)d_ws;
    };
    const float* input = PIN(0);

    const int B = 2, Hin = 224, Win = 224, Cin = 3;
    size_t off = 0;

    bf16_t* xin = (bf16_t*)wsget(d_ws, off, (size_t)B * Hin * Win * Cin * sizeof(bf16_t));
    k_nchw_to_nhwc_bf16<<<gs((size_t)B * Cin * Hin * Win), 256, 0, stream>>>(
        input, xin, B, Cin, Hin, Win);

    // ---- stem conv1: 3 -> 40, 3x3 s2, 112x112
    const int H1 = 112, W1 = 112, C1 = 40;
    const int M1 = B * H1 * W1, K1 = Cin * 9, K1p = padk(K1);
    bf16_t* col1 = (bf16_t*)wsget(d_ws, off, (size_t)M1 * K1p * sizeof(bf16_t));
    k_im2col3x3_bf16<<<gs((size_t)M1 * K1p), 256, 0, stream>>>(
        xin, col1, B, Hin, Win, Cin, 2, 1, H1, W1, K1p);
    bf16_t* w1 = (bf16_t*)wsget(d_ws, off, (size_t)C1 * K1p * sizeof(bf16_t));
    k_f32_to_bf16_rows<<<gs((size_t)C1 * K1p), 256, 0, stream>>>(PIN(49), w1, C1, K1, K1p);
    float* o1 = (float*)wsget(d_ws, off, (size_t)M1 * C1 * sizeof(float));
    gemm(stream, col1, K1p, w1, K1p, o1, nullptr, C1, PIN(50), M1, C1, K1p, 0);
    float* mu = (float*)wsget(d_ws, off, 2048 * sizeof(float));
    float* vv = (float*)wsget(d_ws, off, 2048 * sizeof(float));
    k_channel_stats<<<C1, 256, 0, stream>>>(o1, M1, C1, mu, vv);
    bf16_t* h1 = (bf16_t*)wsget(d_ws, off, (size_t)M1 * C1 * sizeof(bf16_t));
    k_bn_act<<<gs((size_t)M1 * C1), 256, 0, stream>>>(o1, M1, C1, C1, mu, vv,
        PIN(51), PIN(52), 1, nullptr, h1);

    // ---- stem conv2: 40 -> 80, 3x3 s2, 56x56
    const int H2 = 56, W2 = 56, C2 = 80;
    const int M2 = B * H2 * W2, K2 = C1 * 9, K2p = padk(K2);
    bf16_t* col2 = (bf16_t*)wsget(d_ws, off, (size_t)M2 * K2p * sizeof(bf16_t));
    k_im2col3x3_bf16<<<gs((size_t)M2 * K2p), 256, 0, stream>>>(
        h1, col2, B, H1, W1, C1, 2, 1, H2, W2, K2p);
    bf16_t* w2 = (bf16_t*)wsget(d_ws, off, (size_t)C2 * K2p * sizeof(bf16_t));
    k_f32_to_bf16_rows<<<gs((size_t)C2 * K2p), 256, 0, stream>>>(PIN(53), w2, C2, K2, K2p);
    float* o2 = (float*)wsget(d_ws, off, (size_t)M2 * C2 * sizeof(float));
    gemm(stream, col2, K2p, w2, K2p, o2, nullptr, C2, PIN(54), M2, C2, K2p, 0);
    k_channel_stats<<<C2, 256, 0, stream>>>(o2, M2, C2, mu, vv);
    bf16_t* h2 = (bf16_t*)wsget(d_ws, off, (size_t)M2 * C2 * sizeof(bf16_t));
    k_bn_act<<<gs((size_t)M2 * C2), 256, 0, stream>>>(o2, M2, C2, C2, mu, vv,
        PIN(55), PIN(56), 1, nullptr, h2);

    // ---- stem conv3: 80 -> 80, 3x3 s1, + pos_embed
    const int K3 = C2 * 9, K3p = padk(K3);
    bf16_t* col3 = (bf16_t*)wsget(d_ws, off, (size_t)M2 * K3p * sizeof(bf16_t));
    k_im2col3x3_bf16<<<gs((size_t)M2 * K3p), 256, 0, stream>>>(
        h2, col3, B, H2, W2, C2, 1, 1, H2, W2, K3p);
    bf16_t* w3 = (bf16_t*)wsget(d_ws, off, (size_t)C2 * K3p * sizeof(bf16_t));
    k_f32_to_bf16_rows<<<gs((size_t)C2 * K3p), 256, 0, stream>>>(PIN(57), w3, C2, K3, K3p);
    float* x0 = (float*)wsget(d_ws, off, (size_t)M2 * C2 * sizeof(float));
    gemm(stream, col3, K3p, w3, K3p, x0, nullptr, C2, PIN(58), M2, C2, K3p, 0);
    k_channel_stats<<<C2, 256, 0, stream>>>(x0, M2, C2, mu, vv);
    k_bn_act<<<gs((size_t)M2 * C2), 256, 0, stream>>>(x0, M2, C2, C2, mu, vv,
        PIN(59), PIN(60), 0, x0, nullptr);
    k_add_pos_nhwc<<<gs((size_t)M2 * C2), 256, 0, stream>>>(x0, PIN(61), B, H2, W2, C2);

    // ---- grapher (stage 0: C=80, r=4, k=12, dil=1)
    const int Cg = 80, Cgp = padk(Cg), Npix = H2 * W2, rr = 4;
    const int Mcol = (H2 / rr) * (W2 / rr);           // 196
    const int KNN = 12, DIL = 1, KD = KNN * DIL;

    bf16_t* x0b = (bf16_t*)wsget(d_ws, off, (size_t)M2 * Cgp * sizeof(bf16_t));
    k_f32_to_bf16_rows<<<gs((size_t)M2 * Cgp), 256, 0, stream>>>(x0, x0b, M2, Cg, Cgp);
    bf16_t* wf1 = (bf16_t*)wsget(d_ws, off, (size_t)Cg * Cgp * sizeof(bf16_t));
    k_f32_to_bf16_rows<<<gs((size_t)Cg * Cgp), 256, 0, stream>>>(PIN(62), wf1, Cg, Cg, Cgp);
    float* xg = (float*)wsget(d_ws, off, (size_t)M2 * Cg * sizeof(float));
    gemm(stream, x0b, Cgp, wf1, Cgp, xg, nullptr, Cg, PIN(63), M2, Cg, Cgp, 0);
    k_channel_stats<<<Cg, 256, 0, stream>>>(xg, M2, Cg, mu, vv);
    k_bn_act<<<gs((size_t)M2 * Cg), 256, 0, stream>>>(xg, M2, Cg, Cg, mu, vv,
        PIN(64), PIN(65), 0, xg, nullptr);

    float* yg = (float*)wsget(d_ws, off, (size_t)B * Mcol * Cg * sizeof(float));
    k_avgpool_nhwc<<<gs((size_t)B * Mcol * Cg), 256, 0, stream>>>(xg, yg, B, H2, W2, Cg, rr);
    bf16_t* xn = (bf16_t*)wsget(d_ws, off, (size_t)M2 * Cgp * sizeof(bf16_t));
    bf16_t* yn = (bf16_t*)wsget(d_ws, off, (size_t)B * Mcol * Cgp * sizeof(bf16_t));
    k_rownorm_bf16<<<gs(M2), 256, 0, stream>>>(xg, M2, Cg, Cgp, xn);
    k_rownorm_bf16<<<gs(B * Mcol), 256, 0, stream>>>(yg, B * Mcol, Cg, Cgp, yn);

    // grouped-padded z layout: 4 groups of 40 channels padded to 64
    const int GRP = 4, KG = 2 * Cg / GRP /*40*/, GP = padk(KG) /*64*/;
    const int SP = GRP * GP;                           // 256
    float* ip  = (float*)wsget(d_ws, off, (size_t)Npix * Mcol * sizeof(float));
    int*   idx = (int*)wsget(d_ws, off, (size_t)B * Npix * KNN * sizeof(int));
    bf16_t* z  = (bf16_t*)wsget(d_ws, off, (size_t)M2 * SP * sizeof(bf16_t));
    for (int b = 0; b < B; ++b) {
        const bf16_t* xb = xn + (size_t)b * Npix * Cgp;
        const bf16_t* yb = yn + (size_t)b * Mcol * Cgp;
        gemm(stream, xb, Cgp, yb, Cgp, ip, nullptr, Mcol, nullptr, Npix, Mcol, Cgp, 0);
        k_topk_rows<<<gs(Npix), 256, 0, stream>>>(ip, PIN(74), Npix, Mcol, KD, DIL,
                                                  KNN, idx + (size_t)b * Npix * KNN);
        k_build_z<<<gs((size_t)Npix * SP), 256, 0, stream>>>(
            xg + (size_t)b * Npix * Cg, yg + (size_t)b * Mcol * Cg,
            idx + (size_t)b * Npix * KNN, Npix, Cg, KNN, KG, GP, GRP,
            z + (size_t)b * Npix * SP);
    }

    // grouped 1x1 conv 160 -> 160 (groups=4), bn + gelu
    const int C2g = 2 * Cg;                            // 160
    bf16_t* wg = (bf16_t*)wsget(d_ws, off, (size_t)C2g * GP * sizeof(bf16_t));
    k_f32_to_bf16_rows<<<gs((size_t)C2g * GP), 256, 0, stream>>>(PIN(66), wg, C2g, KG, GP);
    float* og = (float*)wsget(d_ws, off, (size_t)M2 * C2g * sizeof(float));
    const int CG = C2g / GRP;                          // 40 outputs per group
    for (int g = 0; g < GRP; ++g) {
        gemm(stream, z + g * GP, SP, wg + (size_t)g * CG * GP, GP,
             og + g * CG, nullptr, C2g, PIN(67) + g * CG, M2, CG, GP, 0);
    }
    k_channel_stats<<<C2g, 256, 0, stream>>>(og, M2, C2g, mu, vv);
    bf16_t* ogb = (bf16_t*)wsget(d_ws, off, (size_t)M2 * C2g * sizeof(bf16_t));
    k_bn_act<<<gs((size_t)M2 * C2g), 256, 0, stream>>>(og, M2, C2g, C2g, mu, vv,
        PIN(68), PIN(69), 2, nullptr, ogb);

    // fc2 1x1: 160 -> 80, bn, + shortcut
    bf16_t* wf2 = (bf16_t*)wsget(d_ws, off, (size_t)Cg * C2g * sizeof(bf16_t));
    k_f32_to_bf16_rows<<<gs((size_t)Cg * C2g), 256, 0, stream>>>(PIN(70), wf2, Cg, C2g, C2g);
    float* g0 = (float*)wsget(d_ws, off, (size_t)M2 * Cg * sizeof(float));
    gemm(stream, ogb, C2g, wf2, C2g, g0, nullptr, Cg, PIN(71), M2, Cg, C2g, 0);
    k_channel_stats<<<Cg, 256, 0, stream>>>(g0, M2, Cg, mu, vv);
    k_bn_act<<<gs((size_t)M2 * Cg), 256, 0, stream>>>(g0, M2, Cg, Cg, mu, vv,
        PIN(72), PIN(73), 0, g0, nullptr);
    k_add_f32<<<gs((size_t)M2 * Cg), 256, 0, stream>>>(g0, x0, g0, (size_t)M2 * Cg);

    // ---- FFN: 80 -> 320 (gelu) -> 80, + residual
    const int Ch = 4 * Cg;
    bf16_t* g0b = (bf16_t*)wsget(d_ws, off, (size_t)M2 * Cgp * sizeof(bf16_t));
    k_f32_to_bf16_rows<<<gs((size_t)M2 * Cgp), 256, 0, stream>>>(g0, g0b, M2, Cg, Cgp);
    bf16_t* wh1 = (bf16_t*)wsget(d_ws, off, (size_t)Ch * Cgp * sizeof(bf16_t));
    k_f32_to_bf16_rows<<<gs((size_t)Ch * Cgp), 256, 0, stream>>>(PIN(75), wh1, Ch, Cg, Cgp);
    float* hmid = (float*)wsget(d_ws, off, (size_t)M2 * Ch * sizeof(float));
    gemm(stream, g0b, Cgp, wh1, Cgp, hmid, nullptr, Ch, PIN(76), M2, Ch, Cgp, 0);
    k_channel_stats<<<Ch, 256, 0, stream>>>(hmid, M2, Ch, mu, vv);
    bf16_t* hmb = (bf16_t*)wsget(d_ws, off, (size_t)M2 * Ch * sizeof(bf16_t));
    k_bn_act<<<gs((size_t)M2 * Ch), 256, 0, stream>>>(hmid, M2, Ch, Ch, mu, vv,
        PIN(77), PIN(78), 2, nullptr, hmb);
    bf16_t* wh2 = (bf16_t*)wsget(d_ws, off, (size_t)Cg * Ch * sizeof(bf16_t));
    k_f32_to_bf16_rows<<<gs((size_t)Cg * Ch), 256, 0, stream>>>(PIN(79), wh2, Cg, Ch, Ch);
    float* f0 = (float*)wsget(d_ws, off, (size_t)M2 * Cg * sizeof(float));
    gemm(stream, hmb, Ch, wh2, Ch, f0, nullptr, Cg, PIN(80), M2, Cg, Ch, 0);
    k_channel_stats<<<Cg, 256, 0, stream>>>(f0, M2, Cg, mu, vv);
    k_bn_act<<<gs((size_t)M2 * Cg), 256, 0, stream>>>(f0, M2, Cg, Cg, mu, vv,
        PIN(81), PIN(82), 0, f0, nullptr);
    k_add_f32<<<gs((size_t)M2 * Cg), 256, 0, stream>>>(f0, g0, f0, (size_t)M2 * Cg);

    // ---- head: 1x1 conv 80 -> 2, upsample x4 to 224, NHWC -> NCHW out
    const int NC = 2;
    bf16_t* f0b = (bf16_t*)wsget(d_ws, off, (size_t)M2 * Cgp * sizeof(bf16_t));
    k_f32_to_bf16_rows<<<gs((size_t)M2 * Cgp), 256, 0, stream>>>(f0, f0b, M2, Cg, Cgp);
    bf16_t* ws_ = (bf16_t*)wsget(d_ws, off, (size_t)NC * Cgp * sizeof(bf16_t));
    k_f32_to_bf16_rows<<<gs((size_t)NC * Cgp), 256, 0, stream>>>(PIN(49), ws_, NC, Cg, Cgp);
    float* s56 = (float*)wsget(d_ws, off, (size_t)M2 * NC * sizeof(float));
    gemm(stream, f0b, Cgp, ws_, Cgp, s56, nullptr, NC, nullptr, M2, NC, Cgp, 0);
    float* s112 = (float*)wsget(d_ws, off, (size_t)B * 112 * 112 * NC * sizeof(float));
    k_up2_bilinear_nhwc<<<gs((size_t)B * 112 * 112 * NC), 256, 0, stream>>>(
        s56, s112, B, 56, 56, NC);
    float* s224 = (float*)wsget(d_ws, off, (size_t)B * 224 * 224 * NC * sizeof(float));
    k_up2_bilinear_nhwc<<<gs((size_t)B * 224 * 224 * NC), 256, 0, stream>>>(
        s112, s224, B, 112, 112, NC);
    k_nhwc_to_nchw_f32<<<gs((size_t)B * NC * 224 * 224), 256, 0, stream>>>(
        s224, (float*)d_out, B, NC, 224, 224);

    (void)in_sizes; (void)out_size; (void)ws_size;
}
